// Generator_74302934221260
// MI455X (gfx1250) — matrix operand order repaired and checked by reading, not compiled, not run on hardware
//
#include <hip/hip_runtime.h>
#include <hip/hip_bf16.h>
#include <math.h>

typedef _Float16 half_t;
typedef __attribute__((ext_vector_type(8)))  _Float16 v8h;
typedef __attribute__((ext_vector_type(16))) _Float16 v16h;
typedef __attribute__((ext_vector_type(8)))  float    v8f;

__device__ __forceinline__ float lrelu(float v) { return v > 0.0f ? v : 0.2f * v; }

#define SHUF16(lo, hi) __builtin_shufflevector(lo, hi, 0,1,2,3,4,5,6,7,8,9,10,11,12,13,14,15)

// ---------------------------------------------------------------------------
// Operand packing kernels (write WMMA fragment-order f16, zero-padded)
// B-frag convention: col n = lane%16 ; slot j -> k = kb*32 + j + (lane>=16?16:0)
// Packed layout: dst[(kb*Ntiles + nt)*512 + lane*16 + j]   (contiguous per lane)
// ---------------------------------------------------------------------------

// Linear weight src[N][K] used as B = src^T
__global__ void k_pack_linw(const float* __restrict__ src, half_t* __restrict__ dst,
                            int N, int K, int Ntiles, int Kblks) {
    int t = blockIdx.x * blockDim.x + threadIdx.x;
    int total = Kblks * Ntiles * 512;
    if (t >= total) return;
    int j    = t & 15;
    int lane = (t >> 4) & 31;
    int nt   = (t >> 9) % Ntiles;
    int kb   = (t >> 9) / Ntiles;
    int n = nt * 16 + (lane & 15);
    int k = kb * 32 + j + ((lane >> 4) ? 16 : 0);
    dst[t] = (n < N && k < K) ? (half_t)src[(size_t)n * K + k] : (half_t)0.0f;
}

// Conv weight src[O][I][3][3] -> per-tap packed B (k = input channel, n = o)
// layout: dst[((r*Kblks + kb)*Ntiles + nt)*512 + lane*16 + j]
__global__ void k_pack_convw(const float* __restrict__ src, half_t* __restrict__ dst,
                             int O, int I, int Ntiles, int Kblks) {
    int t = blockIdx.x * blockDim.x + threadIdx.x;
    int total = 9 * Kblks * Ntiles * 512;
    if (t >= total) return;
    int j    = t & 15;
    int lane = (t >> 4) & 31;
    int q    = t >> 9;                 // q = (r*Kblks + kb)*Ntiles + nt
    int nt = q % Ntiles; q /= Ntiles;
    int kb = q % Kblks;  int r = q / Kblks;
    int o = nt * 16 + (lane & 15);
    int i = kb * 32 + j + ((lane >> 4) ? 16 : 0);
    half_t v = (half_t)0.0f;
    if (o < O && i < I) v = (half_t)src[((size_t)o * I + i) * 9 + r];
    dst[t] = v;
}

__global__ void k_f2h(const float* __restrict__ src, half_t* __restrict__ dst, int n) {
    int t = blockIdx.x * blockDim.x + threadIdx.x;
    if (t < n) dst[t] = (half_t)src[t];
}

// Stage-0 input, NHWC padded: Xmod[b][p][c] = c<3 ? const[c][p]*s0[b][c] : 0
__global__ void k_mod_const(const float* __restrict__ cst, const float* __restrict__ s0,
                            half_t* __restrict__ Xmod, int Cpad) {
    int t = blockIdx.x * blockDim.x + threadIdx.x;
    if (t >= 32 * 16 * Cpad) return;
    int c = t % Cpad; int q = t / Cpad;
    int p = q & 15;   int b = q >> 4;
    half_t v = (half_t)0.0f;
    if (c < 3) v = (half_t)(cst[c * 16 + p] * s0[b * 3 + c]);
    Xmod[t] = v;
}

// Bilinear 2x upsample (half-pixel, edge clamp) + next-stage modulation.
// Yin: NHWC [B][H][W][C] f32 ; Xmod: NHWC [B][2H][2W][Cpad] f16 (pad zeroed)
__global__ void k_up2_mod(const float* __restrict__ Yin, const float* __restrict__ s,
                          half_t* __restrict__ Xmod, int Bc, int C, int Cpad,
                          int H, int W) {
    int OH = 2 * H, OW = 2 * W;
    size_t total = (size_t)Bc * OH * OW * Cpad;
    size_t t = (size_t)blockIdx.x * blockDim.x + threadIdx.x;
    if (t >= total) return;
    int c = (int)(t % Cpad); size_t q = t / Cpad;
    int ox = (int)(q % OW);  q /= OW;
    int oy = (int)(q % OH);  int b = (int)(q / OH);
    half_t out = (half_t)0.0f;
    if (c < C) {
        float sy = (oy + 0.5f) * 0.5f - 0.5f;
        float sx = (ox + 0.5f) * 0.5f - 0.5f;
        int y0 = (int)floorf(sy); float fy = sy - (float)y0;
        int x0 = (int)floorf(sx); float fx = sx - (float)x0;
        int y0c = min(max(y0, 0), H - 1), y1c = min(max(y0 + 1, 0), H - 1);
        int x0c = min(max(x0, 0), W - 1), x1c = min(max(x0 + 1, 0), W - 1);
        const float* base = Yin + (size_t)b * H * W * C;
        float v00 = base[((size_t)y0c * W + x0c) * C + c];
        float v01 = base[((size_t)y0c * W + x1c) * C + c];
        float v10 = base[((size_t)y1c * W + x0c) * C + c];
        float v11 = base[((size_t)y1c * W + x1c) * C + c];
        float v = (1.0f - fy) * ((1.0f - fx) * v00 + fx * v01) +
                  fy          * ((1.0f - fx) * v10 + fx * v11);
        out = (half_t)(v * s[(size_t)b * C + c]);
    }
    Xmod[t] = out;
}

// ---------------------------------------------------------------------------
// WMMA GEMM: D = A[32,K] x B + bias ; B pre-packed in fragment order.
// Each wave computes a 16M x (16*NTG)N tile: A-fragment reused NTG times.
// mode 0: leaky, f16 out (mapping).  mode 1: f32 out (modulation).
// ---------------------------------------------------------------------------
template <int NTG>
__global__ __launch_bounds__(32)
void k_gemm_wmma(const half_t* __restrict__ A, const half_t* __restrict__ Bpk,
                 const float* __restrict__ bias, half_t* __restrict__ outH,
                 float* __restrict__ outF, int K, int N, int Ntiles, int mode) {
    const int lane = threadIdx.x;
    const int row  = lane & 15;
    const int hi   = lane >> 4;
    const int mBase  = blockIdx.y * 16;
    const int ntBase = blockIdx.x * NTG;

    const half_t* Arow = A + (size_t)(mBase + row) * K + hi * 16;
    const half_t* Bp   = Bpk + (size_t)ntBase * 512 + lane * 16;
    const size_t  Bstr = (size_t)Ntiles * 512;

    const v8f zf = {};
    v8f acc[NTG];
#pragma unroll
    for (int g = 0; g < NTG; ++g) acc[g] = zf;

    for (int kb = 0; kb < K / 32; ++kb) {
        v8h alo = *(const v8h*)(Arow + kb * 32);
        v8h ahi = *(const v8h*)(Arow + kb * 32 + 8);
        v16h a = SHUF16(alo, ahi);
#pragma unroll
        for (int g = 0; g < NTG; ++g) {
            v8h blo = *(const v8h*)(Bp + g * 512);
            v8h bhi = *(const v8h*)(Bp + g * 512 + 8);
            v16h b = SHUF16(blo, bhi);
            acc[g] = __builtin_amdgcn_wmma_f32_16x16x32_f16(false, a, false, b,
                                                            (short)0, acc[g], false, false);
        }
        Bp += Bstr;
    }
#pragma unroll
    for (int g = 0; g < NTG; ++g) {
        int n = (ntBase + g) * 16 + row;
        if (n < N) {
            float bs = bias[n];
#pragma unroll
            for (int r = 0; r < 8; ++r) {
                int m = mBase + (hi ? r + 8 : r);
                float v = acc[g][r] + bs;
                if (mode == 0) outH[(size_t)m * N + n] = (half_t)lrelu(v);
                else           outF[(size_t)m * N + n] = v;
            }
        }
    }
}

// ---------------------------------------------------------------------------
// Modulated 3x3 SAME conv as 9 shifted channel-GEMMs (NHWC, WMMA).
//   X   : NHWC f16 [B][H][W][I]  (I padded to mult of 32, pad channels zero)
//   Wpk : packed per-tap fragments [((r*Kblks+kb)*Ntiles+nt)*512 + lane*16 + j]
// Each wave: 16 positions x (16*NTG) output channels (A reused NTG times).
// Epilogue: acc*demod[o] + nw*noise(NCHW); leaky->Y(NHWC f32) or tanh->d_out(NCHW).
// ---------------------------------------------------------------------------
template <int NTG>
__global__ __launch_bounds__(32)
void k_conv_wmma(const half_t* __restrict__ X, const half_t* __restrict__ Wpk,
                 const float* __restrict__ demod, const float* __restrict__ noise,
                 const float* __restrict__ nwp, int stage,
                 float* __restrict__ Yout, float* __restrict__ finalOut,
                 int I, int O, int Ntiles, int H, int W, int isLast) {
    const int lane = threadIdx.x;
    const int row  = lane & 15;
    const int hi   = lane >> 4;
    const int mBase  = blockIdx.x * 16;
    const int ntBase = blockIdx.y * NTG;
    const int HW = H * W;
    const int Kblks = I / 32;

    const int mA = mBase + row;
    const int bA = mA / HW;
    const int pA = mA - bA * HW;
    const int yA = pA / W;
    const int xA = pA - yA * W;

    const half_t* Bp0 = Wpk + (size_t)ntBase * 512 + lane * 16;
    const v8h vz = {};
    const v8f zf = {};
    v8f acc[NTG];
#pragma unroll
    for (int g = 0; g < NTG; ++g) acc[g] = zf;

#pragma unroll
    for (int r = 0; r < 9; ++r) {
        const int ky = r / 3, kx = r - (r / 3) * 3;
        int iy = yA + ky - 1, ix = xA + kx - 1;
        bool valid = ((unsigned)iy < (unsigned)H) && ((unsigned)ix < (unsigned)W);
        int iyc = min(max(iy, 0), H - 1);
        int ixc = min(max(ix, 0), W - 1);
        const half_t* Arow = X + ((size_t)(bA * H + iyc) * W + ixc) * I + hi * 16;
        const half_t* Bp = Bp0 + (size_t)(r * Kblks) * Ntiles * 512;
        for (int kb = 0; kb < Kblks; ++kb) {
            v8h alo = *(const v8h*)(Arow + kb * 32);
            v8h ahi = *(const v8h*)(Arow + kb * 32 + 8);
            if (!valid) { alo = vz; ahi = vz; }     // cndmask, no divergence
            v16h a = SHUF16(alo, ahi);
#pragma unroll
            for (int g = 0; g < NTG; ++g) {
                v8h blo = *(const v8h*)(Bp + g * 512);
                v8h bhi = *(const v8h*)(Bp + g * 512 + 8);
                v16h b = SHUF16(blo, bhi);
                acc[g] = __builtin_amdgcn_wmma_f32_16x16x32_f16(false, a, false, b,
                                                                (short)0, acc[g], false, false);
            }
            Bp += (size_t)Ntiles * 512;
        }
    }

    const float nw = nwp[stage];
#pragma unroll
    for (int g = 0; g < NTG; ++g) {
        int o = (ntBase + g) * 16 + row;
        if (o < O) {
            const float dm = demod[o];
#pragma unroll
            for (int r = 0; r < 8; ++r) {
                int m = mBase + (hi ? r + 8 : r);
                int b2 = m / HW;
                int p  = m - b2 * HW;
                size_t nidx = ((size_t)b2 * O + o) * HW + p;      // NCHW (noise, d_out)
                float v = acc[g][r] * dm + nw * noise[nidx];
                if (isLast) finalOut[nidx] = tanhf(v);
                else        Yout[((size_t)b2 * HW + p) * O + o] = lrelu(v);  // NHWC
            }
        }
    }
}

// ---------------------------------------------------------------------------
// Host orchestration
// ---------------------------------------------------------------------------
extern "C" void kernel_launch(void* const* d_in, const int* in_sizes, int n_in,
                              void* d_out, int out_size, void* d_ws, size_t ws_size,
                              hipStream_t stream) {
    (void)in_sizes; (void)n_in; (void)out_size; (void)ws_size;

    const float* z     = (const float*)d_in[0];
    const float* map_w = (const float*)d_in[1];
    const float* map_b = (const float*)d_in[2];
    const float* cst   = (const float*)d_in[3];
    const float* conv_w[6]; for (int i = 0; i < 6; ++i) conv_w[i] = (const float*)d_in[4 + i];
    const float* mod_w[6];  for (int i = 0; i < 6; ++i) mod_w[i]  = (const float*)d_in[10 + i];
    const float* mod_b[6];  for (int i = 0; i < 6; ++i) mod_b[i]  = (const float*)d_in[16 + i];
    const float* demod[6];  for (int i = 0; i < 6; ++i) demod[i]  = (const float*)d_in[22 + i];
    const float* nwp   = (const float*)d_in[28];
    const float* noise[6];  for (int i = 0; i < 6; ++i) noise[i]  = (const float*)d_in[29 + i];
    float* outp = (float*)d_out;

    static const int I_[6]  = {3, 512, 512, 256, 128, 64};
    static const int O_[6]  = {512, 512, 256, 128, 64, 3};
    static const int R_[6]  = {4, 8, 16, 32, 64, 128};
    static const int IP_[6] = {32, 512, 512, 256, 128, 64};   // I padded to 32
    const int B = 32;
    const int TB = 256;

    char* wsb = (char*)d_ws; size_t off = 0;
    auto alloc = [&](size_t bytes) -> char* {
        char* p = wsb + off; off = (off + bytes + 255) & ~(size_t)255; return p;
    };

    // packed weights
    half_t* mapWpk = (half_t*)alloc(8ull * 16 * 32 * 512 * sizeof(half_t));
    half_t* modWpk[6]; int modNt[6];
    for (int i = 0; i < 6; ++i) {
        modNt[i] = (I_[i] + 15) / 16;
        modWpk[i] = (half_t*)alloc((size_t)16 * modNt[i] * 512 * sizeof(half_t));
    }
    half_t* convWpk[6]; int convNt[6], convKb[6];
    for (int i = 0; i < 6; ++i) {
        convNt[i] = (O_[i] + 15) / 16;
        convKb[i] = IP_[i] / 32;
        convWpk[i] = (half_t*)alloc((size_t)9 * convKb[i] * convNt[i] * 512 * sizeof(half_t));
    }
    half_t* stylesA = (half_t*)alloc((size_t)B * 512 * sizeof(half_t));
    half_t* stylesB = (half_t*)alloc((size_t)B * 512 * sizeof(half_t));
    float* sbuf[6];
    for (int i = 0; i < 6; ++i) sbuf[i] = (float*)alloc((size_t)B * I_[i] * sizeof(float));
    half_t* Xmod = (half_t*)alloc((size_t)B * 128 * 128 * 64 * sizeof(half_t)); // max NHWC input
    float*  Ybuf = (float*) alloc((size_t)B * 64 * 64 * 64 * sizeof(float));    // max NHWC output

    // ---- weight packing ----
    for (int l = 0; l < 8; ++l) {
        int tot = 16 * 32 * 512;
        k_pack_linw<<<(tot + TB - 1) / TB, TB, 0, stream>>>(
            map_w + (size_t)l * 512 * 512, mapWpk + (size_t)l * tot, 512, 512, 32, 16);
    }
    for (int i = 0; i < 6; ++i) {
        int tot = 16 * modNt[i] * 512;
        k_pack_linw<<<(tot + TB - 1) / TB, TB, 0, stream>>>(
            mod_w[i], modWpk[i], I_[i], 512, modNt[i], 16);
    }
    for (int i = 0; i < 6; ++i) {
        int tot = 9 * convKb[i] * convNt[i] * 512;
        k_pack_convw<<<(tot + TB - 1) / TB, TB, 0, stream>>>(
            conv_w[i], convWpk[i], O_[i], I_[i], convNt[i], convKb[i]);
    }
    k_f2h<<<(B * 512 + TB - 1) / TB, TB, 0, stream>>>(z, stylesA, B * 512);

    // ---- mapping network: 8 x (GEMM + bias + leaky), 16x64 wave tiles ----
    half_t* cur = stylesA; half_t* nxt = stylesB;
    for (int l = 0; l < 8; ++l) {
        dim3 g(32 / 4, 2);
        k_gemm_wmma<4><<<g, 32, 0, stream>>>(cur, mapWpk + (size_t)l * 16 * 32 * 512,
                                             map_b + (size_t)l * 512, nxt, nullptr,
                                             512, 512, 32, 0);
        half_t* t = cur; cur = nxt; nxt = t;
    }

    // ---- style modulation GEMMs ----
    for (int i = 0; i < 6; ++i) {
        if (modNt[i] % 4 == 0) {
            dim3 g(modNt[i] / 4, 2);
            k_gemm_wmma<4><<<g, 32, 0, stream>>>(cur, modWpk[i], mod_b[i], nullptr, sbuf[i],
                                                 512, I_[i], modNt[i], 1);
        } else {
            dim3 g(modNt[i], 2);
            k_gemm_wmma<1><<<g, 32, 0, stream>>>(cur, modWpk[i], mod_b[i], nullptr, sbuf[i],
                                                 512, I_[i], modNt[i], 1);
        }
    }

    // ---- stage-0 input (NHWC padded) ----
    k_mod_const<<<(B * 16 * IP_[0] + TB - 1) / TB, TB, 0, stream>>>(cst, sbuf[0], Xmod, IP_[0]);

    // ---- synthesis ----
    for (int i = 0; i < 6; ++i) {
        int H = R_[i];
        int Mtiles = (B * H * H) / 16;
        int isLast = (i == 5);
        if (convNt[i] % 4 == 0) {
            dim3 g((unsigned)Mtiles, (unsigned)(convNt[i] / 4));
            k_conv_wmma<4><<<g, 32, 0, stream>>>(Xmod, convWpk[i], demod[i], noise[i], nwp, i,
                                                 Ybuf, outp, IP_[i], O_[i], convNt[i], H, H, isLast);
        } else {
            dim3 g((unsigned)Mtiles, (unsigned)convNt[i]);
            k_conv_wmma<1><<<g, 32, 0, stream>>>(Xmod, convWpk[i], demod[i], noise[i], nwp, i,
                                                 Ybuf, outp, IP_[i], O_[i], convNt[i], H, H, isLast);
        }
        if (i < 5) {
            size_t tot = (size_t)B * (2 * H) * (2 * H) * IP_[i + 1];
            k_up2_mod<<<(unsigned)((tot + TB - 1) / TB), TB, 0, stream>>>(
                Ybuf, sbuf[i + 1], Xmod, B, O_[i], IP_[i + 1], H, H);
        }
    }
}
